// PPO_7318624273070
// MI455X (gfx1250) — compile-verified, matrix operand order checked by reading, not run on hardware
//
#include <hip/hip_runtime.h>
#include <hip/hip_bf16.h>

// ---------------------------------------------------------------------------
// Problem constants: B=4, N=2048, F_IN=64, HID=256, HEADS=4, D=64
// ---------------------------------------------------------------------------
#define NB      4
#define NN      2048
#define ROWS    (NB * NN)      // 8192
#define HID     256
#define HEADS   4
#define DHEAD   64
#define MAXDEG  192

typedef __attribute__((ext_vector_type(4)))  _Float16 v4h;
typedef __attribute__((ext_vector_type(8)))  _Float16 v8h;
typedef __attribute__((ext_vector_type(16))) _Float16 v16h;
typedef __attribute__((ext_vector_type(8)))  float    v8f;

// ---------------------------------------------------------------------------
// WMMA GEMM, compile-time K/NC. A: f16 row-major [M][K] (activations, f16
// shadow copy). Bt: f16 [NC][K] (pre-transposed weights). C: f32 [M][NC],
// optional f16 shadow C16. 256 thr = 8 waves; 64(M)x32(N) tile; K-step 32.
// A tile staged in LDS (one b128 global load + one b128 ds store per thread,
// software-pipelined); B operands loaded straight from global (2 x b128/lane,
// weight rows are L2-resident). Lane layout per CDNA5 ISA 7.12.2.
// ---------------------------------------------------------------------------
template<int K, int NC>
__global__ __launch_bounds__(256) void gemm_wmma(
    const _Float16* __restrict__ A, const _Float16* __restrict__ Bt,
    const float* __restrict__ bias, float* __restrict__ C,
    _Float16* __restrict__ C16, int relu)
{
    __shared__ _Float16 As[64][40];          // padded rows: 80 B, 16B-aligned

    const int tileM = blockIdx.x * 64;
    const int tileN = blockIdx.y * 32;
    const int tid   = threadIdx.x;
    const int lane  = tid & 31;
    const int w     = tid >> 5;
    const int wm    = (w & 3) * 16;          // 4 waves along M
    const int wn    = (w >> 2) * 16;         // 2 waves along N

    const int kb = ((lane >> 4) & 1) * 8;    // upper half-wave -> K offset 8
    const int mm = wm + (lane & 15);
    const int nn = wn + (lane & 15);

    // staging assignment: thread -> (row, 8-half chunk); 256*8 = 64*32 exact
    const int sr = tid >> 2;
    const int sc = (tid & 3) * 8;
    const _Float16* aGlob = A + (size_t)(tileM + sr) * K + sc;
    const _Float16* bRow  = Bt + (size_t)(tileN + nn) * K;

    v8f acc = {};
    v8h areg = *(const v8h*)aGlob;           // prefetch first A tile

#pragma unroll
    for (int k0 = 0; k0 < K; k0 += 32) {
        *(v8h*)&As[sr][sc] = areg;           // ds_store_b128
        __syncthreads();
        if (k0 + 32 < K)                     // pipeline next tile past barrier
            areg = *(const v8h*)(aGlob + k0 + 32);

        v8h b0 = *(const v8h*)(bRow + k0 + kb);        // global b128
        v8h b1 = *(const v8h*)(bRow + k0 + 16 + kb);
        v8h a0 = *(const v8h*)&As[mm][kb];             // ds_load_b128
        v8h a1 = *(const v8h*)&As[mm][16 + kb];

        v16h av = __builtin_shufflevector(a0, a1, 0,1,2,3,4,5,6,7,8,9,10,11,12,13,14,15);
        v16h bv = __builtin_shufflevector(b0, b1, 0,1,2,3,4,5,6,7,8,9,10,11,12,13,14,15);
        acc = __builtin_amdgcn_wmma_f32_16x16x32_f16(
            false, av, false, bv, (short)0, acc, false, false);
        __syncthreads();
    }

    const int col  = tileN + wn + (lane & 15);
    const int mrow = tileM + wm + ((lane & 16) ? 8 : 0);
    const float bb = bias ? bias[col] : 0.f;
#pragma unroll
    for (int r = 0; r < 8; ++r) {
        float v = acc[r] + bb;
        if (relu) v = fmaxf(v, 0.f);
        C[(size_t)(mrow + r) * NC + col] = v;
        if (C16) C16[(size_t)(mrow + r) * NC + col] = (_Float16)v;
    }
}

// ---------------------------------------------------------------------------
// One-time weight transpose+convert: W[K][Nc] f32 -> Wt[Nc][K] f16
// ---------------------------------------------------------------------------
__global__ __launch_bounds__(256) void wt_kernel(
    const float* __restrict__ W, _Float16* __restrict__ Wt, int K, int Nc)
{
    const int idx = blockIdx.x * 256 + threadIdx.x;
    if (idx >= K * Nc) return;
    const int k = idx / Nc, n = idx % Nc;
    Wt[(size_t)n * K + k] = (_Float16)W[idx];
}

// f32 -> f16 copy (row-major), float4 in / 8B out
__global__ __launch_bounds__(256) void cvt16_kernel(
    const float* __restrict__ X, _Float16* __restrict__ Y)
{
    const int idx = blockIdx.x * 256 + threadIdx.x;       // per float4
    float4 x = ((const float4*)X)[idx];
    v4h y = { (_Float16)x.x, (_Float16)x.y, (_Float16)x.z, (_Float16)x.w };
    *(v4h*)(Y + (size_t)idx * 4) = y;
}

// ---------------------------------------------------------------------------
// Adjacency compaction: one wave per row. Two independent float4 loads per
// iteration (1 KB in flight per wave) + prefetch; ballot+prefix compaction,
// fixed ballot sequence -> deterministic. This is the 67 MB roofline kernel.
// ---------------------------------------------------------------------------
__global__ __launch_bounds__(256) void build_adj(
    const float* __restrict__ A, unsigned short* __restrict__ adj,
    int* __restrict__ deg)
{
    const int row  = blockIdx.x * 8 + (threadIdx.x >> 5);
    const int lane = threadIdx.x & 31;
    const float* arow = A + (size_t)row * NN;
    const unsigned lt = (1u << lane) - 1u;
    unsigned short* ap = adj + (size_t)row * MAXDEG;
    int base = 0;

    for (int c0 = 0; c0 < NN; c0 += 256) {
        float4 x0 = *(const float4*)&arow[c0 + lane * 4];
        float4 x1 = *(const float4*)&arow[c0 + 128 + lane * 4];
        __builtin_prefetch(&arow[c0 + 256 + lane * 4], 0, 1);
        const float e0[8] = { x0.x, x0.y, x0.z, x0.w, x1.x, x1.y, x1.z, x1.w };
#pragma unroll
        for (int j = 0; j < 8; ++j) {
            const float v = e0[j];
            const int col = c0 + ((j >> 2) * 128) + lane * 4 + (j & 3);
            const unsigned msk = (unsigned)__ballot(v != 0.f);
            const int pre = __popc(msk & lt);
            if (v != 0.f) {
                const int pos = base + pre;
                if (pos < MAXDEG) ap[pos] = (unsigned short)col;
            }
            base += __popc(msk);
        }
    }
    if (lane == 0) deg[row] = base < MAXDEG ? base : MAXDEG;
}

// ---------------------------------------------------------------------------
// e_src/e_dst: per (row, head) dot of Xt head-slice with a_src / a_dst.
// ---------------------------------------------------------------------------
__global__ __launch_bounds__(256) void esd_kernel(
    const float* __restrict__ Xt, const float* __restrict__ asrc,
    const float* __restrict__ adst, float* __restrict__ e_src,
    float* __restrict__ e_dst)
{
    const int idx = blockIdx.x * blockDim.x + threadIdx.x;  // row*4 + h
    const int row = idx >> 2, hh = idx & 3;
    const float4* xp = (const float4*)(Xt + (size_t)row * HID + hh * DHEAD);
    const float4* sp = (const float4*)(asrc + hh * DHEAD);
    const float4* dp = (const float4*)(adst + hh * DHEAD);
    float s = 0.f, d = 0.f;
#pragma unroll 4
    for (int i = 0; i < 16; ++i) {
        float4 x = xp[i], a = sp[i], b = dp[i];
        s += x.x * a.x + x.y * a.y + x.z * a.z + x.w * a.w;
        d += x.x * b.x + x.y * b.y + x.z * b.z + x.w * b.w;
    }
    e_src[idx] = s;
    e_dst[idx] = d;
}

// ---------------------------------------------------------------------------
// Diagonal of masked softmax: wave per row, online softmax over neighbors.
// ---------------------------------------------------------------------------
__global__ __launch_bounds__(256) void diag_kernel(
    const unsigned short* __restrict__ adj, const int* __restrict__ deg,
    const float* __restrict__ e_src, const float* __restrict__ e_dst,
    float* __restrict__ diag)
{
    const int row  = blockIdx.x * 8 + (threadIdx.x >> 5);
    const int lane = threadIdx.x & 31;
    const int b    = row >> 11;
    const int d    = deg[row];
    const unsigned short* ap = adj + (size_t)row * MAXDEG;

    for (int hh = 0; hh < HEADS; ++hh) {
        const float s = e_src[row * HEADS + hh];
        float m = -1e30f, sum = 0.f;
        for (int j = lane; j < d; j += 32) {
            const int col = ap[j];
            float v = s + e_dst[((size_t)(b * NN + col)) * HEADS + hh];
            v = v > 0.f ? v : 0.2f * v;                  // leaky_relu(0.2)
            const float mn = fmaxf(m, v);
            sum = sum * __expf(m - mn) + __expf(v - mn);
            m = mn;
        }
        for (int off = 16; off > 0; off >>= 1) {
            const float mo = __shfl_xor(m, off);
            const float so = __shfl_xor(sum, off);
            const float mn = fmaxf(m, mo);
            sum = sum * __expf(m - mn) + so * __expf(mo - mn);
            m = mn;
        }
        if (lane == 0) {
            float vi = s + e_dst[(size_t)row * HEADS + hh];
            vi = vi > 0.f ? vi : 0.2f * vi;
            diag[row * HEADS + hh] = __expf(vi - m) / sum;
        }
    }
}

// h <- diag (per head) * Xt ; also emits f16 shadow for the next GEMM's A.
__global__ __launch_bounds__(256) void scale_kernel(
    const float* __restrict__ Xt, const float* __restrict__ diag,
    float* __restrict__ H, _Float16* __restrict__ H16)
{
    const int idx = blockIdx.x * blockDim.x + threadIdx.x;   // float4 index
    const int row = idx >> 6;
    const int c4  = idx & 63;
    const float dg = diag[row * HEADS + (c4 >> 4)];
    float4 x = ((const float4*)Xt)[idx];
    float4 o = make_float4(x.x * dg, x.y * dg, x.z * dg, x.w * dg);
    ((float4*)H)[idx] = o;
    v4h o16 = { (_Float16)o.x, (_Float16)o.y, (_Float16)o.z, (_Float16)o.w };
    *(v4h*)(H16 + (size_t)idx * 4) = o16;
}

// actor head: relu(a3 . aw4 + ab4) - mask*1e8
__global__ __launch_bounds__(256) void logits_kernel(
    const float* __restrict__ A3, const float* __restrict__ w4,
    const float* __restrict__ b4, const float* __restrict__ mask,
    float* __restrict__ logits)
{
    const int row = blockIdx.x * blockDim.x + threadIdx.x;
    const float4* xp = (const float4*)(A3 + (size_t)row * 64);
    const float4* wp = (const float4*)w4;
    float s = 0.f;
#pragma unroll
    for (int i = 0; i < 16; ++i) {
        float4 x = xp[i], w0 = wp[i];
        s += x.x * w0.x + x.y * w0.y + x.z * w0.z + x.w * w0.w;
    }
    s = fmaxf(s + b4[0], 0.f);
    logits[row] = s - mask[row] * 1e8f;
}

// per-batch softmax over N=2048
__global__ __launch_bounds__(256) void softmaxN_kernel(
    const float* __restrict__ logits, float* __restrict__ out)
{
    __shared__ float red[256];
    const int b = blockIdx.x, t = threadIdx.x;
    const float* x = logits + b * NN;
    float m = -1e30f;
    for (int i = t; i < NN; i += 256) m = fmaxf(m, x[i]);
    red[t] = m; __syncthreads();
    for (int s = 128; s > 0; s >>= 1) {
        if (t < s) red[t] = fmaxf(red[t], red[t + s]);
        __syncthreads();
    }
    m = red[0]; __syncthreads();
    float sum = 0.f;
    for (int i = t; i < NN; i += 256) sum += __expf(x[i] - m);
    red[t] = sum; __syncthreads();
    for (int s = 128; s > 0; s >>= 1) {
        if (t < s) red[t] += red[t + s];
        __syncthreads();
    }
    const float inv = 1.f / red[0];
    for (int i = t; i < NN; i += 256) out[b * NN + i] = __expf(x[i] - m) * inv;
}

// g[b][c] = mean_n h[b][n][c]
__global__ __launch_bounds__(256) void mean_kernel(
    const float* __restrict__ H, float* __restrict__ g)
{
    const int b = blockIdx.x, c = threadIdx.x;
    const float* hp = H + (size_t)b * NN * HID + c;
    float s = 0.f;
#pragma unroll 8
    for (int n = 0; n < NN; ++n) s += hp[(size_t)n * HID];
    g[b * HID + c] = s * (1.f / (float)NN);
}

// critic MLP, one block: 4x256 -> 4x64 -> 4x32 -> 4x16 -> 4
__global__ __launch_bounds__(256) void critic_kernel(
    const float* __restrict__ g,
    const float* __restrict__ cw1, const float* __restrict__ cb1,
    const float* __restrict__ cw2, const float* __restrict__ cb2,
    const float* __restrict__ cw3, const float* __restrict__ cb3,
    const float* __restrict__ cw4, const float* __restrict__ cb4,
    float* __restrict__ out)
{
    __shared__ float v0[NB * 256], v1[NB * 64], v2[NB * 32], v3[NB * 16];
    const int t = threadIdx.x;
    for (int i = t; i < NB * 256; i += 256) v0[i] = g[i];
    __syncthreads();
    {
        const int b = t >> 6, j = t & 63;
        float s = cb1[j];
        for (int k = 0; k < 256; ++k) s += v0[b * 256 + k] * cw1[k * 64 + j];
        v1[t] = fmaxf(s, 0.f);
    }
    __syncthreads();
    if (t < NB * 32) {
        const int b = t >> 5, j = t & 31;
        float s = cb2[j];
        for (int k = 0; k < 64; ++k) s += v1[b * 64 + k] * cw2[k * 32 + j];
        v2[t] = fmaxf(s, 0.f);
    }
    __syncthreads();
    if (t < NB * 16) {
        const int b = t >> 4, j = t & 15;
        float s = cb3[j];
        for (int k = 0; k < 32; ++k) s += v2[b * 32 + k] * cw3[k * 16 + j];
        v3[t] = fmaxf(s, 0.f);
    }
    __syncthreads();
    if (t < NB) {
        float s = cb4[0];
        for (int k = 0; k < 16; ++k) s += v3[t * 16 + k] * cw4[k];
        out[t] = fmaxf(s, 0.f);
    }
}

// ---------------------------------------------------------------------------
extern "C" void kernel_launch(void* const* d_in, const int* in_sizes, int n_in,
                              void* d_out, int out_size, void* d_ws, size_t ws_size,
                              hipStream_t stream)
{
    const float* X        = (const float*)d_in[0];
    const float* A        = (const float*)d_in[1];
    const float* mask     = (const float*)d_in[2];
    const float* emb_W    = (const float*)d_in[3];
    const float* emb_b    = (const float*)d_in[4];
    const float* gat_W    = (const float*)d_in[5];
    const float* gat_asrc = (const float*)d_in[6];
    const float* gat_adst = (const float*)d_in[7];
    const float* aw1 = (const float*)d_in[8];
    const float* ab1 = (const float*)d_in[9];
    const float* aw2 = (const float*)d_in[10];
    const float* ab2 = (const float*)d_in[11];
    const float* aw3 = (const float*)d_in[12];
    const float* ab3 = (const float*)d_in[13];
    const float* aw4 = (const float*)d_in[14];
    const float* ab4 = (const float*)d_in[15];
    const float* cw1 = (const float*)d_in[16];
    const float* cb1 = (const float*)d_in[17];
    const float* cw2 = (const float*)d_in[18];
    const float* cb2 = (const float*)d_in[19];
    const float* cw3 = (const float*)d_in[20];
    const float* cb3 = (const float*)d_in[21];
    const float* cw4 = (const float*)d_in[22];
    const float* cb4 = (const float*)d_in[23];
    float* out = (float*)d_out;

    // ---- workspace layout (bytes) ----
    char* w = (char*)d_ws;
    float* h      = (float*)w;   w += (size_t)ROWS * HID * 4;     // 8 MB
    float* Xt     = (float*)w;   w += (size_t)ROWS * HID * 4;     // 8 MB (reused: a1,a2 f32)
    float* a3     = (float*)w;   w += (size_t)ROWS * 64 * 4;      // 2 MB
    float* e_src  = (float*)w;   w += (size_t)ROWS * HEADS * 4;
    float* e_dst  = (float*)w;   w += (size_t)ROWS * HEADS * 4;
    float* diag   = (float*)w;   w += (size_t)ROWS * HEADS * 4;
    float* logits = (float*)w;   w += (size_t)ROWS * 4;
    float* g      = (float*)w;   w += (size_t)NB * HID * 4;
    int*   deg    = (int*)w;     w += (size_t)ROWS * 4;
    unsigned short* adj = (unsigned short*)w;  w += (size_t)ROWS * MAXDEG * 2;  // 3 MB
    _Float16* X16   = (_Float16*)w;  w += (size_t)ROWS * 64 * 2;   // 1 MB
    _Float16* h16   = (_Float16*)w;  w += (size_t)ROWS * HID * 2;  // 4 MB
    _Float16* a1_16 = (_Float16*)w;  w += (size_t)ROWS * 128 * 2;  // 2 MB
    _Float16* a2_16 = (_Float16*)w;  w += (size_t)ROWS * 128 * 2;  // 2 MB
    _Float16* embWt = (_Float16*)w;  w += (size_t)64 * HID * 2;
    _Float16* gatWt = (_Float16*)w;  w += (size_t)3 * HID * HID * 2;
    _Float16* aw1t  = (_Float16*)w;  w += (size_t)HID * 128 * 2;
    _Float16* aw2t  = (_Float16*)w;  w += (size_t)128 * 128 * 2;
    _Float16* aw3t  = (_Float16*)w;  w += (size_t)128 * 64 * 2;

    // 0) one-time f16 prep: weights transposed, X converted
    wt_kernel<<<(64 * HID + 255) / 256, 256, 0, stream>>>(emb_W, embWt, 64, HID);
    for (int l = 0; l < 3; ++l)
        wt_kernel<<<(HID * HID + 255) / 256, 256, 0, stream>>>(
            gat_W + (size_t)l * HID * HID, gatWt + (size_t)l * HID * HID, HID, HID);
    wt_kernel<<<(HID * 128 + 255) / 256, 256, 0, stream>>>(aw1, aw1t, HID, 128);
    wt_kernel<<<(128 * 128 + 255) / 256, 256, 0, stream>>>(aw2, aw2t, 128, 128);
    wt_kernel<<<(128 * 64 + 255) / 256, 256, 0, stream>>>(aw3, aw3t, 128, 64);
    cvt16_kernel<<<(ROWS * 64 / 4) / 256, 256, 0, stream>>>(X, X16);

    // 1) compress A once (the single 67 MB read; the roofline floor)
    build_adj<<<ROWS / 8, 256, 0, stream>>>(A, adj, deg);

    // 2) embedding GEMM: (8192x64)@(64x256)+b  -> h (+h16)
    gemm_wmma<64, HID><<<dim3(ROWS / 64, HID / 32), 256, 0, stream>>>(
        X16, embWt, emb_b, h, h16, 0);

    // 3) three GAT layers (diagonal-only attention)
    for (int l = 0; l < 3; ++l) {
        gemm_wmma<HID, HID><<<dim3(ROWS / 64, HID / 32), 256, 0, stream>>>(
            h16, gatWt + (size_t)l * HID * HID, nullptr, Xt, nullptr, 0);
        esd_kernel<<<(ROWS * HEADS) / 256, 256, 0, stream>>>(
            Xt, gat_asrc + (size_t)l * HEADS * DHEAD,
            gat_adst + (size_t)l * HEADS * DHEAD, e_src, e_dst);
        diag_kernel<<<ROWS / 8, 256, 0, stream>>>(adj, deg, e_src, e_dst, diag);
        scale_kernel<<<(ROWS * HID / 4) / 256, 256, 0, stream>>>(Xt, diag, h, h16);
    }

    // 4) actor MLP (f32 a1/a2 reuse Xt; f16 shadows feed the next GEMM)
    float* a1 = Xt;
    float* a2 = Xt + (size_t)ROWS * 128;
    gemm_wmma<HID, 128><<<dim3(ROWS / 64, 128 / 32), 256, 0, stream>>>(
        h16, aw1t, ab1, a1, a1_16, 1);
    gemm_wmma<128, 128><<<dim3(ROWS / 64, 128 / 32), 256, 0, stream>>>(
        a1_16, aw2t, ab2, a2, a2_16, 1);
    gemm_wmma<128, 64><<<dim3(ROWS / 64, 64 / 32), 256, 0, stream>>>(
        a2_16, aw3t, ab3, a3, nullptr, 1);
    logits_kernel<<<ROWS / 256, 256, 0, stream>>>(a3, aw4, ab4, mask, logits);
    softmaxN_kernel<<<NB, 256, 0, stream>>>(logits, out);

    // 5) critic
    mean_kernel<<<NB, 256, 0, stream>>>(h, g);
    critic_kernel<<<1, 256, 0, stream>>>(g, cw1, cb1, cw2, cb2, cw3, cb3,
                                         cw4, cb4, out + ROWS);
}
